// LSTM_60189671686761
// MI455X (gfx1250) — matrix operand order repaired and checked by reading, not compiled, not run on hardware
//
#include <hip/hip_runtime.h>
#include <math.h>

typedef __attribute__((ext_vector_type(16))) _Float16 v16h;
typedef __attribute__((ext_vector_type(8)))  _Float16 v8h;
typedef __attribute__((ext_vector_type(8)))  float    v8f;

#define B_ 16
#define S_ 256
#define V_ 32000
#define H_ 32
#define NT_ (V_ / 16)   // 2000 N-tiles
#define MT_ ((B_ * S_) / 16) // 256 M-tiles

// ---------------------------------------------------------------------------
// Kernel 1: fp32 -> f16 convert (embedding matrix, 1.024M elems -> 2 MB)
// ---------------------------------------------------------------------------
__global__ __launch_bounds__(256) void cvt_f32_f16(const float* __restrict__ src,
                                                   _Float16* __restrict__ dst, int n) {
  int i = blockIdx.x * blockDim.x + threadIdx.x;
  if (i < n) dst[i] = (_Float16)src[i];
}

// ---------------------------------------------------------------------------
// Kernel 2: sequential LayerNorm-LSTM scan. 1 block per batch, 128 threads
// (one gate-row each). W row lives in registers; x||h lives in LDS and is
// read as float4 broadcasts. Each wave keeps a private copy of c and
// redundantly writes identical h/x values to LDS so only 2 barriers/step
// are needed.
// ---------------------------------------------------------------------------
__global__ __launch_bounds__(128) void lstm_scan(
    const int*   __restrict__ tokens,   // [B, S]
    const float* __restrict__ hidden,   // [B, 1, H]
    const float* __restrict__ cell,     // [B, 1, H]
    const float* __restrict__ embed_w,  // [V, H]
    const float* __restrict__ W,        // [4H, 2H] row-major
    const float* __restrict__ bias,     // [4H]
    const float* __restrict__ ln_w,     // [4, H] flat
    const float* __restrict__ ln_b,     // [4, H] flat
    _Float16*    __restrict__ outs_h)   // [B*S, H] f16
{
  __shared__ __align__(16) float data[64];  // [x_t(32) | h(32)]
  __shared__ float act[128];
  __shared__ float redS[4], redQ[4];

  const int b    = blockIdx.x;
  const int r    = threadIdx.x;       // gate-row 0..127; gate = r>>5, j = r&31
  const int lane = r & 31;
  const int wid  = r >> 5;

  // W row -> registers (fully unrolled so it stays in VGPRs)
  float w[64];
  #pragma unroll
  for (int k = 0; k < 64; ++k) w[k] = W[r * 64 + k];
  const float brow = bias[r];
  const float lw = ln_w[r], lb = ln_b[r];

  // every wave keeps its own copy of c[lane]
  float c_reg = cell[b * H_ + lane];

  if (r < 32) {
    data[32 + r] = hidden[b * H_ + r];
    data[r]      = embed_w[(size_t)tokens[b * S_] * H_ + r];
  }
  __syncthreads();

  const float4* d4 = (const float4*)data;

  for (int t = 0; t < S_; ++t) {
    // ---- gate pre-activation: dot(W[r,:], [x;h]) ----
    float g = brow;
    #pragma unroll
    for (int k4 = 0; k4 < 16; ++k4) {
      float4 v = d4[k4];
      g += w[4*k4+0]*v.x + w[4*k4+1]*v.y + w[4*k4+2]*v.z + w[4*k4+3]*v.w;
    }

    // ---- LayerNorm stats over all 128 gate values ----
    float s = g, q = g * g;
    #pragma unroll
    for (int m = 16; m >= 1; m >>= 1) {
      s += __shfl_xor(s, m, 32);
      q += __shfl_xor(q, m, 32);
    }
    if (lane == 0) { redS[wid] = s; redQ[wid] = q; }
    __syncthreads();                                    // B1
    float sum = redS[0] + redS[1] + redS[2] + redS[3];
    float sq  = redQ[0] + redQ[1] + redQ[2] + redQ[3];
    float mu  = sum * (1.0f / 128.0f);
    float var = sq * (1.0f / 128.0f) - mu * mu;
    float gn  = (g - mu) * rsqrtf(var + 1e-5f) * lw + lb;

    // ---- activations: gates 0/1/3 sigmoid, gate 2 exact GELU ----
    float a;
    if (wid == 2) a = 0.5f * gn * (1.0f + erff(gn * 0.70710678118654752f));
    else          a = 1.0f / (1.0f + expf(-gn));
    act[r] = a;
    __syncthreads();                                    // B2

    // ---- state update (every wave redundantly, bit-identical writes) ----
    {
      float gi = act[lane], gf = act[32 + lane], gh = act[64 + lane], go = act[96 + lane];
      c_reg   = gf * c_reg + gi * gh;
      float h = go * c_reg;
      data[32 + lane] = h;
      if (t + 1 < S_)
        data[lane] = embed_w[(size_t)tokens[b * S_ + t + 1] * H_ + lane];
      if (wid == 0)
        outs_h[((size_t)b * S_ + t) * H_ + lane] = (_Float16)h;
    }
    // no barrier: all waves write identical values before re-reading
  }
}

// ---------------------------------------------------------------------------
// Kernel 3: tied output projection via WMMA.
// logits[4096, 32000] = outs[4096,32] @ embed[32000,32]^T + out_b
// K == 32 exactly: one v_wmma_f32_16x16x32_f16 per 16x16 tile.
// One wave per tile; 8 waves/block; 64000 blocks = 512000 tiles.
// ---------------------------------------------------------------------------
__global__ __launch_bounds__(256) void tied_proj(
    const _Float16* __restrict__ Ah,    // [4096, 32]
    const _Float16* __restrict__ Eh,    // [32000, 32]
    const float*    __restrict__ out_b, // [32000]
    float*          __restrict__ out)   // [4096, 32000]
{
  const int wid  = threadIdx.x >> 5;
  const int lane = threadIdx.x & 31;
  const int tile = blockIdx.x * 8 + wid;        // 0 .. 511999
  const int tmi  = tile / NT_;                  // M-tile index
  const int tni  = tile - tmi * NT_;            // N-tile index
  const int tileM = tmi << 4;
  const int tileN = tni << 4;
  const int l  = lane & 15;
  const int hi = lane >> 4;                     // lane half

  // A (16x32, f16): lane l<16 holds M=l, K 0-7 then 16-23;
  //                 lane l+16 holds M=l, K 8-15 then 24-31.
  union { v16h v; v8h h[2]; } a, bm;
  const _Float16* arow = Ah + (size_t)(tileM + l) * H_ + hi * 8;
  a.h[0] = *(const v8h*)(arow);
  a.h[1] = *(const v8h*)(arow + 16);

  // B (32x16, f16): lane l<16 holds column N=l, K 0-7 then 16-23;
  //                 lane l+16 holds column N=l, K 8-15 then 24-31.
  const _Float16* brow = Eh + (size_t)(tileN + l) * H_ + hi * 8;
  bm.h[0] = *(const v8h*)(brow);
  bm.h[1] = *(const v8h*)(brow + 16);

  // bias through the accumulator: C/D column = lane&15 in every VGPR
  float bb = out_b[tileN + l];
  v8f c;
  #pragma unroll
  for (int i = 0; i < 8; ++i) c[i] = bb;

  v8f d = __builtin_amdgcn_wmma_f32_16x16x32_f16(
      /*neg_a=*/false, a.v, /*neg_b=*/false, bm.v,
      /*c_mod=*/(short)0, c, /*reuse_a=*/false, /*reuse_b=*/false);

  // D layout: VGPR i -> row tileM + i (+8 for upper half-lanes), col lane&15
  float* orow = out + (size_t)(tileM + 8 * hi) * V_ + tileN + l;
  #pragma unroll
  for (int i = 0; i < 8; ++i) orow[(size_t)i * V_] = d[i];
}

// ---------------------------------------------------------------------------
extern "C" void kernel_launch(void* const* d_in, const int* in_sizes, int n_in,
                              void* d_out, int out_size, void* d_ws, size_t ws_size,
                              hipStream_t stream) {
  const int*   tokens  = (const int*)  d_in[0];
  const float* hidden  = (const float*)d_in[1];
  const float* cell    = (const float*)d_in[2];
  const float* embed_w = (const float*)d_in[3];
  const float* W       = (const float*)d_in[4];
  const float* bvec    = (const float*)d_in[5];
  const float* ln_w    = (const float*)d_in[6];
  const float* ln_b    = (const float*)d_in[7];
  const float* out_b   = (const float*)d_in[8];
  float*       out     = (float*)d_out;

  _Float16* Eh = (_Float16*)d_ws;                                   // 32000*32 f16 = 2.0 MB
  _Float16* Ah = (_Float16*)((char*)d_ws + (size_t)V_ * H_ * 2);    // 4096*32  f16 = 256 KB

  const int nE = V_ * H_;
  cvt_f32_f16<<<(nE + 255) / 256, 256, 0, stream>>>(embed_w, Eh, nE);
  lstm_scan<<<B_, 128, 0, stream>>>(tokens, hidden, cell, embed_w, W, bvec,
                                    ln_w, ln_b, Ah);
  tied_proj<<<(MT_ * NT_) / 8, 256, 0, stream>>>(Ah, Eh, out_b, out);
}
